// unCAT_6803228197703
// MI455X (gfx1250) — matrix operand order repaired and checked by reading, not compiled, not run on hardware
//
#include <hip/hip_runtime.h>

// GAT forward on gfx1250 (MI455X), fused flash-style softmax + bf16 WMMA.
// Inputs: x[8192,512] f32, adj[8192,8192] i32, W[512,128] f32, a_i[128] f32, a_j[128] f32
// Output: h_prime[8192,128] f32
//
// Roofline: adj (256 MB) read exactly twice (row stats + fused attention GEMM)
// = 512 MB -> ~22 us at 23.3 TB/s. attention@h = 17.2 GFLOP in bf16 WMMA.
// Probabilities are computed once per element (shared across the block's 8
// waves through LDS), so exp work is ~134M transcendentals total.

#define N_NODES 8192
#define DI      512
#define DO      128
#define LEAK    0.2f
#define NEG_BIG (-9.0e15f)

typedef __attribute__((ext_vector_type(16))) __bf16 v16bf;
typedef __attribute__((ext_vector_type(8)))  float  v8f;

union Frag16 { uint4 u[2]; v16bf v; };   // 32 bytes = one 16-bit A/B operand

// ---------------- conversions ----------------
__global__ void k_conv_x(const float* __restrict__ x, __bf16* __restrict__ xb, int n) {
  int i = blockIdx.x * blockDim.x + threadIdx.x;
  if (i < n) xb[i] = (__bf16)x[i];
}

// W[k,n] -> WbT[n,k]  (column-major so a WMMA B-column is contiguous)
__global__ void k_conv_w(const float* __restrict__ W, __bf16* __restrict__ WbT) {
  int i = blockIdx.x * blockDim.x + threadIdx.x;
  if (i < DI * DO) {
    int k = i / DO, n = i % DO;
    WbT[(size_t)n * DI + k] = (__bf16)W[i];
  }
}

// ---------------- h = x @ W  (bf16 WMMA, fp32 accum) ----------------
// grid: 512 blocks x 256 threads; block = 16 rows, wave w = 16-col tile.
__global__ void k_gemm_h(const __bf16* __restrict__ xb, const __bf16* __restrict__ WbT,
                         float* __restrict__ h, __bf16* __restrict__ hbT) {
  const int lane = threadIdx.x & 31;
  const int wave = threadIdx.x >> 5;
  const int m0 = blockIdx.x * 16;
  const int n0 = wave * 16;
  const int r  = lane & 15;    // A row / B col within tile
  const int kh = lane >> 4;    // lane-half selects K sub-stripe

  // A: row m0+r, segments at K = kh*8 and 16+kh*8 (ISA 16-bit A layout)
  const __bf16* ap = xb  + (size_t)(m0 + r) * DI + kh * 8;
  // B: col n0+r, segments at K = kh*8 and 16+kh*8 (matches A)
  const __bf16* bp = WbT + (size_t)(n0 + r) * DI + kh * 8;

  v8f acc = {};
  #pragma unroll 4
  for (int k0 = 0; k0 < DI; k0 += 32) {
    Frag16 a, b;
    a.u[0] = *(const uint4*)(ap + k0);
    a.u[1] = *(const uint4*)(ap + k0 + 16);
    b.u[0] = *(const uint4*)(bp + k0);
    b.u[1] = *(const uint4*)(bp + k0 + 16);
    acc = __builtin_amdgcn_wmma_f32_16x16x32_bf16(false, a.v, false, b.v,
                                                  (short)0, acc, false, false);
  }
  // C layout: VGPR v -> M = v + 8*(lane>=16), N = lane&15
  const int rb = kh * 8;
  #pragma unroll
  for (int v = 0; v < 8; ++v) {
    float val = acc[v];
    int row = m0 + rb + v;
    h[(size_t)row * DO + n0 + r] = val;
    hbT[(size_t)(n0 + r) * N_NODES + row] = (__bf16)val;   // contiguous -> one b128 store
  }
}

// ---------------- ei = h@a_i, ej = h@a_j  (wave per row) ----------------
__global__ void k_eiej(const float* __restrict__ h, const float* __restrict__ a_i,
                       const float* __restrict__ a_j, float* __restrict__ ei,
                       float* __restrict__ ej) {
  const int lane = threadIdx.x & 31;
  const int row  = blockIdx.x * 8 + (threadIdx.x >> 5);
  const float* hr = h + (size_t)row * DO;
  float si = 0.f, sj = 0.f;
  #pragma unroll
  for (int t = 0; t < 4; ++t) {
    float v = hr[lane + t * 32];
    si += v * a_i[lane + t * 32];
    sj += v * a_j[lane + t * 32];
  }
  #pragma unroll
  for (int off = 16; off >= 1; off >>= 1) {
    si += __shfl_xor(si, off, 32);
    sj += __shfl_xor(sj, off, 32);
  }
  if (lane == 0) { ei[row] = si; ej[row] = sj; }
}

// ---------------- row softmax stats (m, 1/Z) — single adj sweep ----------------
__global__ void k_rowstats(const int* __restrict__ adj, const float* __restrict__ ei,
                           const float* __restrict__ ej, float* __restrict__ rowm,
                           float* __restrict__ rinv) {
  const int lane = threadIdx.x & 31;
  const int row  = blockIdx.x * 8 + (threadIdx.x >> 5);
  const float eir = ei[row];
  const int* ar = adj + (size_t)row * N_NODES;
  float m = -__builtin_inff();
  float s = 0.f;
  for (int c = 0; c < N_NODES / 128; ++c) {
    int j = c * 128 + lane * 4;
    if (j + 1024 < N_NODES) __builtin_prefetch(ar + j + 1024, 0, 0);  // global_prefetch_b8
    int4   a4 = *(const int4*)(ar + j);
    float4 e4 = *(const float4*)(ej + j);
    int   av[4] = {a4.x, a4.y, a4.z, a4.w};
    float ev[4] = {e4.x, e4.y, e4.z, e4.w};
    #pragma unroll
    for (int q = 0; q < 4; ++q) {
      float t = eir + ev[q];
      float e = t > 0.f ? t : LEAK * t;
      e = av[q] > 0 ? e : NEG_BIG;          // masked -> -9e15, matches reference
      if (e > m) { s = s * __expf(m - e) + 1.f; m = e; }   // online rescale (rare branch)
      else       { s += __expf(e - m); }
    }
  }
  #pragma unroll
  for (int off = 16; off >= 1; off >>= 1) {
    float mo = __shfl_xor(m, off, 32);
    float so = __shfl_xor(s, off, 32);
    float nm = fmaxf(m, mo);
    s = s * __expf(m - nm) + so * __expf(mo - nm);
    m = nm;
  }
  if (lane == 0) { rowm[row] = m; rinv[row] = 1.0f / s; }
}

// ---------------- fused attention @ h  (bf16 WMMA) ----------------
// Block = 16 rows x 128 cols (8 waves, wave w owns col tile w*16).
// Per 64-wide j chunk: block computes a 16x64 prob tile into a DOUBLE-BUFFERED
// LDS tile (one exp per element, shared by all 8 waves; b128 adj loads), ONE
// barrier, then each wave ds-gathers two A fragments and global-loads two B
// fragments from L2-resident hbT -> two v_wmma per wave per barrier.
// Double-buffer safety: reads of buf[p] finish before barrier i+1 (enforced by
// s_wait_dscnt inside __syncthreads); rewrites of buf[p] happen after it.
__global__ void k_attn(const int* __restrict__ adj, const float* __restrict__ ei,
                       const float* __restrict__ ej, const float* __restrict__ rowm,
                       const float* __restrict__ rinv, const __bf16* __restrict__ hbT,
                       float* __restrict__ out) {
  __shared__ __bf16 pTile[2][16 * 64];
  __shared__ float s_ei[16], s_m[16], s_iv[16];

  const int tid  = threadIdx.x;
  const int lane = tid & 31;
  const int wave = tid >> 5;
  const int rowbase = blockIdx.x * 16;
  const int n0 = wave * 16;

  if (tid < 16) {
    int i = rowbase + tid;
    s_ei[tid] = ei[i];
    s_m[tid]  = rowm[i];
    s_iv[tid] = rinv[i];
  }
  __syncthreads();

  // probability-tile producer coords: 4 elements per thread, e = tid*4
  const int pr = tid >> 4;              // tile row  (e >> 6)
  const int pk = (tid * 4) & 63;        // chunk col (multiple of 4)
  const int* arow = adj + (size_t)(rowbase + pr) * N_NODES + pk;
  const float mi = s_m[pr], iv = s_iv[pr], eiv = s_ei[pr];

  // WMMA fragment coords (ISA 16-bit A layout: lane<16 -> K kh*8..+7, 16+kh*8..+7)
  const int fr = lane & 15;
  const int kh = lane >> 4;
  const __bf16* bbase = hbT + (size_t)(n0 + fr) * N_NODES + kh * 8;

  v8f acc = {};
  for (int it = 0; it < N_NODES / 64; ++it) {
    const int j0 = it * 64;
    __bf16* buf = pTile[it & 1];
    // ---- producer: 4 probabilities per thread ----
    {
      if (j0 + 512 < N_NODES) __builtin_prefetch(arow + j0 + 512, 0, 0);
      int4   a4 = *(const int4*)(arow + j0);
      float4 e4 = *(const float4*)(ej + j0 + pk);
      int   av[4] = {a4.x, a4.y, a4.z, a4.w};
      float ev[4] = {e4.x, e4.y, e4.z, e4.w};
      union { __bf16 b[4]; uint2 u; } pkd;
      #pragma unroll
      for (int q = 0; q < 4; ++q) {
        float t = eiv + ev[q];
        t = t > 0.f ? t : LEAK * t;
        t = av[q] > 0 ? t : NEG_BIG;
        pkd.b[q] = (__bf16)(__expf(t - mi) * iv);  // masked -> 0; all-masked row -> uniform
      }
      ((uint2*)buf)[tid] = pkd.u;                  // ds_store_b64
    }
    __syncthreads();                               // single barrier per 2 WMMAs
    // ---- consumers: two K=32 WMMAs ----
    const __bf16* abase = buf + fr * 64;
    Frag16 a, b;
    a.u[0] = *(const uint4*)(abase + kh * 8);            // ds_load_b128
    a.u[1] = *(const uint4*)(abase + 16 + kh * 8);
    b.u[0] = *(const uint4*)(bbase + j0);                // global_load_b128
    b.u[1] = *(const uint4*)(bbase + j0 + 16);
    acc = __builtin_amdgcn_wmma_f32_16x16x32_bf16(false, a.v, false, b.v,
                                                  (short)0, acc, false, false);
    a.u[0] = *(const uint4*)(abase + 32 + kh * 8);
    a.u[1] = *(const uint4*)(abase + 48 + kh * 8);
    b.u[0] = *(const uint4*)(bbase + j0 + 32);
    b.u[1] = *(const uint4*)(bbase + j0 + 48);
    acc = __builtin_amdgcn_wmma_f32_16x16x32_bf16(false, a.v, false, b.v,
                                                  (short)0, acc, false, false);
  }
  const int rb = kh * 8;
  #pragma unroll
  for (int v = 0; v < 8; ++v)
    out[(size_t)(rowbase + rb + v) * DO + n0 + fr] = acc[v];
}

// ---------------- launcher ----------------
extern "C" void kernel_launch(void* const* d_in, const int* in_sizes, int n_in,
                              void* d_out, int out_size, void* d_ws, size_t ws_size,
                              hipStream_t stream) {
  (void)in_sizes; (void)n_in; (void)out_size; (void)ws_size;
  const float* x   = (const float*)d_in[0];
  const int*   adj = (const int*)d_in[1];
  const float* W   = (const float*)d_in[2];
  const float* a_i = (const float*)d_in[3];
  const float* a_j = (const float*)d_in[4];
  float* out = (float*)d_out;

  char* p = (char*)d_ws;
  auto carve = [&](size_t bytes) -> char* {
    char* r = p; p += (bytes + 255) & ~(size_t)255; return r;
  };
  __bf16* xb   = (__bf16*)carve((size_t)N_NODES * DI * sizeof(__bf16));   // 8 MB
  __bf16* WbT  = (__bf16*)carve((size_t)DI * DO * sizeof(__bf16));        // 128 KB
  float*  h    = (float*) carve((size_t)N_NODES * DO * sizeof(float));    // 4 MB
  __bf16* hbT  = (__bf16*)carve((size_t)DO * N_NODES * sizeof(__bf16));   // 2 MB
  float*  ei   = (float*) carve((size_t)N_NODES * sizeof(float));
  float*  ej   = (float*) carve((size_t)N_NODES * sizeof(float));
  float*  rowm = (float*) carve((size_t)N_NODES * sizeof(float));
  float*  rinv = (float*) carve((size_t)N_NODES * sizeof(float));

  k_conv_x  <<<(N_NODES * DI + 255) / 256, 256, 0, stream>>>(x, xb, N_NODES * DI);
  k_conv_w  <<<(DI * DO + 255) / 256,      256, 0, stream>>>(W, WbT);
  k_gemm_h  <<<N_NODES / 16,               256, 0, stream>>>(xb, WbT, h, hbT);
  k_eiej    <<<N_NODES / 8,                256, 0, stream>>>(h, a_i, a_j, ei, ej);
  k_rowstats<<<N_NODES / 8,                256, 0, stream>>>(adj, ei, ej, rowm, rinv);
  k_attn    <<<N_NODES / 16,               256, 0, stream>>>(adj, ei, ej, rowm, rinv, hbT, out);
}
